// StationGAT_35459249996284
// MI455X (gfx1250) — matrix operand order repaired and checked by reading, not compiled, not run on hardware
//
#include <hip/hip_runtime.h>
#include <hip/hip_bf16.h>
#include <math.h>

typedef __attribute__((ext_vector_type(16))) __bf16 bf16x16;
typedef __attribute__((ext_vector_type(4)))  __bf16 bf16x4;
typedef __attribute__((ext_vector_type(8)))  float  f32x8;

#define TILE_K 32
#define LDS_PAD 8

// ---------------------------------------------------------------------------
// Tiled bf16 WMMA GEMM: C[M,Nc] = A[M,K] * B[K,Nc]  (A,B f32 in memory,
// converted to bf16 in LDS; f32 accumulate). Optional fused bias + ReLU.
// Block tile = (WM*MF*16) x (WN*NF*16); each of the 8 waves computes an
// (MF*16) x (NF*16) sub-tile. Requires K%32==0, Nc%BN==0, 16B-aligned rows;
// M guarded by zero padding. Software-pipelined global prefetch.
// ---------------------------------------------------------------------------
template <int MF, int NF, int WM, int WN>
__global__ __launch_bounds__(256)
void gemm_bf16_wmma(const float* __restrict__ A, const float* __restrict__ B,
                    const float* __restrict__ bias, float* __restrict__ C,
                    int M, int K, int Nc, int addBias, int doRelu)
{
    constexpr int BM = WM * MF * 16;          // 128
    constexpr int BN = WN * NF * 16;          // 128 or 64
    static_assert(WM * WN == 8, "8 waves per block");
    static_assert(BM == 128, "staging assumes BM==128");

    __shared__ __bf16 lA [BM][TILE_K + LDS_PAD];
    __shared__ __bf16 lBt[BN][TILE_K + LDS_PAD];   // B transposed: [n][k]

    const int tid   = threadIdx.x;
    const int lane  = tid & 31;
    const int wv    = tid >> 5;               // 0..7
    const int wm    = wv % WM;
    const int wn    = wv / WM;
    const int mBase = blockIdx.x * BM;
    const int nBase = blockIdx.y * BN;

    // per-thread fixed staging slots
    const int aq  = tid & 7,  ar0 = tid >> 3;     // A: k-quad aq, rows ar0+32j (j<4)
    constexpr int QPR  = BN / 4;                  // n-quads per B row
    constexpr int KST  = 256 / QPR;               // k rows covered per pass
    constexpr int JB   = TILE_K / KST;            // B staging iterations
    const int bq  = tid % QPR, bk0 = tid / QPR;   // B: n-quad bq, k = bk0+KST*j

    f32x8 acc[MF][NF] = {};
    float4 ra[4], rb[JB];

    // prologue: prefetch first tiles
    #pragma unroll
    for (int j = 0; j < 4; ++j) {
        int gm = mBase + ar0 + 32 * j;
        ra[j] = make_float4(0.f, 0.f, 0.f, 0.f);
        if (gm < M) ra[j] = *(const float4*)&A[(size_t)gm * K + aq * 4];
    }
    #pragma unroll
    for (int j = 0; j < JB; ++j)
        rb[j] = *(const float4*)&B[(size_t)(bk0 + KST * j) * Nc + nBase + bq * 4];

    for (int kk = 0; kk < K; kk += TILE_K) {
        // --- stage registers -> LDS (bf16)
        #pragma unroll
        for (int j = 0; j < 4; ++j) {
            bf16x4 p;
            p[0] = (__bf16)ra[j].x; p[1] = (__bf16)ra[j].y;
            p[2] = (__bf16)ra[j].z; p[3] = (__bf16)ra[j].w;
            *(bf16x4*)&lA[ar0 + 32 * j][aq * 4] = p;     // 8B store, aligned
        }
        #pragma unroll
        for (int j = 0; j < JB; ++j) {
            int kr = bk0 + KST * j;
            lBt[bq * 4 + 0][kr] = (__bf16)rb[j].x;
            lBt[bq * 4 + 1][kr] = (__bf16)rb[j].y;
            lBt[bq * 4 + 2][kr] = (__bf16)rb[j].z;
            lBt[bq * 4 + 3][kr] = (__bf16)rb[j].w;
        }
        __syncthreads();

        // --- prefetch next K-step while WMMAs run
        const int kn = kk + TILE_K;
        if (kn < K) {
            #pragma unroll
            for (int j = 0; j < 4; ++j) {
                int gm = mBase + ar0 + 32 * j;
                ra[j] = make_float4(0.f, 0.f, 0.f, 0.f);
                if (gm < M) ra[j] = *(const float4*)&A[(size_t)gm * K + kn + aq * 4];
            }
            #pragma unroll
            for (int j = 0; j < JB; ++j)
                rb[j] = *(const float4*)&B[(size_t)(kn + bk0 + KST * j) * Nc
                                           + nBase + bq * 4];
        }

        // --- A fragments (MF 16x32 slabs), wave32 layout:
        // lanes 0-15: K 0..7 / 16..23 ; lanes 16-31: K 8..15 / 24..31
        bf16x16 fa[MF];
        const int kb = (lane >= 16) ? 8 : 0;
        #pragma unroll
        for (int mf = 0; mf < MF; ++mf) {
            const __bf16* ap = &lA[wm * MF * 16 + mf * 16 + (lane & 15)][0];
            #pragma unroll
            for (int i = 0; i < 8; ++i) {
                fa[mf][i]     = ap[kb + i];
                fa[mf][i + 8] = ap[kb + 16 + i];
            }
        }
        // --- B fragments: col = lane&15; lanes 0-15 K 0..7 / 16..23, lanes 16-31 K 8..15 / 24..31
        const int ksb = (lane >= 16) ? 8 : 0;
        #pragma unroll
        for (int nf = 0; nf < NF; ++nf) {
            bf16x16 fb;
            const __bf16* bp = &lBt[wn * NF * 16 + nf * 16 + (lane & 15)][ksb];
            #pragma unroll
            for (int i = 0; i < 8; ++i) { fb[i] = bp[i]; fb[i + 8] = bp[16 + i]; }
            #pragma unroll
            for (int mf = 0; mf < MF; ++mf)
                acc[mf][nf] = __builtin_amdgcn_wmma_f32_16x16x32_bf16(
                    false, fa[mf], false, fb, (short)0, acc[mf][nf], false, false);
        }
        __syncthreads();
    }

    // --- store (C/D layout: col = lane&15; row = r + 8*(lane>=16))
    #pragma unroll
    for (int mf = 0; mf < MF; ++mf) {
        #pragma unroll
        for (int nf = 0; nf < NF; ++nf) {
            int col = nBase + wn * NF * 16 + nf * 16 + (lane & 15);
            #pragma unroll
            for (int r = 0; r < 8; ++r) {
                int m = mBase + wm * MF * 16 + mf * 16 + r + ((lane >= 16) ? 8 : 0);
                if (m < M) {
                    float v = acc[mf][nf][r];
                    if (addBias) v += bias[col];
                    if (doRelu)  v = fmaxf(v, 0.0f);
                    C[(size_t)m * Nc + col] = v;
                }
            }
        }
    }
}

// ---------------------------------------------------------------------------
// Per-(node,head) attention dots: alpha_src/dst[n,h] = sum_c h[n,h,c]*a[h,c]
// One wave per (n,h); C fixed at 128 (4 f32 per lane), shuffle reduction.
// ---------------------------------------------------------------------------
__global__ __launch_bounds__(256)
void node_alpha(const float* __restrict__ h, const float* __restrict__ a_s,
                const float* __restrict__ a_d, int N, int H,
                float* __restrict__ asrc, float* __restrict__ adst)
{
    int wv   = (blockIdx.x * blockDim.x + threadIdx.x) >> 5;
    int lane = threadIdx.x & 31;
    if (wv >= N * H) return;
    int n = wv / H, hd = wv % H;
    const float* hp  = h   + (size_t)n * H * 128 + hd * 128;
    const float* asp = a_s + hd * 128;
    const float* adp = a_d + hd * 128;
    float ss = 0.f, sd = 0.f;
    #pragma unroll
    for (int j = 0; j < 4; ++j) {
        float v = hp[lane + 32 * j];
        ss += v * asp[lane + 32 * j];
        sd += v * adp[lane + 32 * j];
    }
    #pragma unroll
    for (int off = 16; off > 0; off >>= 1) {
        ss += __shfl_down(ss, off, 32);
        sd += __shfl_down(sd, off, 32);
    }
    if (lane == 0) { asrc[wv] = ss; adst[wv] = sd; }
}

__global__ void fill_f32(float* __restrict__ p, float v, int n)
{
    int i = blockIdx.x * blockDim.x + threadIdx.x;
    if (i < n) p[i] = v;
}

// float atomic max via signed-max / unsigned-min trick
__device__ inline void atomicMaxFloat(float* addr, float val)
{
    if (val >= 0.0f) atomicMax((int*)addr, __float_as_int(val));
    else             atomicMin((unsigned int*)addr, __float_as_uint(val));
}

// pass 1: e = leaky_relu(a_src[src] + a_dst[dst]); segment max into m[dst]
__global__ __launch_bounds__(256)
void edge_logits(const float* __restrict__ asrc, const float* __restrict__ adst,
                 const int* __restrict__ srcI, const int* __restrict__ dstI,
                 int E0, int Etot, int H, float* __restrict__ ebuf,
                 float* __restrict__ m)
{
    int i = blockIdx.x * blockDim.x + threadIdx.x;
    if (i >= Etot * H) return;
    int e = i / H, hd = i % H;
    int s = (e < E0) ? srcI[e] : (e - E0);
    int d = (e < E0) ? dstI[e] : (e - E0);
    float v = asrc[s * H + hd] + adst[d * H + hd];
    v = (v > 0.0f) ? v : 0.2f * v;          // leaky_relu slope 0.2
    ebuf[i] = v;
    atomicMaxFloat(&m[d * H + hd], v);
}

// pass 2: ex = exp(e - m[dst]); segment sum into denom[dst]
__global__ __launch_bounds__(256)
void edge_norm(const int* __restrict__ dstI, int E0, int Etot, int H,
               float* __restrict__ ebuf, const float* __restrict__ m,
               float* __restrict__ denom)
{
    int i = blockIdx.x * blockDim.x + threadIdx.x;
    if (i >= Etot * H) return;
    int e = i / H, hd = i % H;
    int d = (e < E0) ? dstI[e] : (e - E0);
    float ex = __expf(ebuf[i] - m[d * H + hd]);
    ebuf[i] = ex;
    atomicAdd(&denom[d * H + hd], ex);
}

// pass 3: out[dst] += alpha * h[src]; one wave per (edge, head), coalesced
__global__ __launch_bounds__(256)
void edge_aggregate(const float* __restrict__ h, const float* __restrict__ ex,
                    const float* __restrict__ denom,
                    const int* __restrict__ srcI, const int* __restrict__ dstI,
                    int E0, int Etot, int H, float* __restrict__ out)
{
    int wv   = (blockIdx.x * blockDim.x + threadIdx.x) >> 5;
    int lane = threadIdx.x & 31;
    if (wv >= Etot * H) return;
    int e = wv / H, hd = wv % H;
    int s = (e < E0) ? srcI[e] : (e - E0);
    int d = (e < E0) ? dstI[e] : (e - E0);
    float alpha = ex[e * H + hd] / (denom[d * H + hd] + 1e-16f);
    const float* hp = h   + (size_t)s * H * 128 + hd * 128;
    float*       op = out + (size_t)d * H * 128 + hd * 128;
    #pragma unroll
    for (int j = 0; j < 4; ++j) {
        int c = lane + 32 * j;
        atomicAdd(&op[c], alpha * hp[c]);
    }
}

__global__ __launch_bounds__(256)
void bias_relu(float* __restrict__ x, const float* __restrict__ b, int total, int F)
{
    int i = blockIdx.x * blockDim.x + threadIdx.x;
    if (i >= total) return;
    x[i] = fmaxf(x[i] + b[i % F], 0.0f);
}

// final 64 -> 4 projection (tiny)
__global__ __launch_bounds__(256)
void head2_k(const float* __restrict__ hin, const float* __restrict__ W,
             const float* __restrict__ b, int N, float* __restrict__ out)
{
    int i = blockIdx.x * blockDim.x + threadIdx.x;
    if (i >= N * 4) return;
    int n = i >> 2, j = i & 3;
    float s = b[j];
    #pragma unroll
    for (int k = 0; k < 64; ++k) s += hin[n * 64 + k] * W[k * 4 + j];
    out[i] = s;
}

// ---------------------------------------------------------------------------
extern "C" void kernel_launch(void* const* d_in, const int* in_sizes, int n_in,
                              void* d_out, int out_size, void* d_ws, size_t ws_size,
                              hipStream_t stream)
{
    (void)n_in; (void)out_size; (void)ws_size;

    const float* x   = (const float*)d_in[0];
    const int*   ei  = (const int*)  d_in[1];
    const float* W0  = (const float*)d_in[2];
    const float* as0 = (const float*)d_in[3];
    const float* ad0 = (const float*)d_in[4];
    const float* b0  = (const float*)d_in[5];
    const float* W1  = (const float*)d_in[6];
    const float* as1 = (const float*)d_in[7];
    const float* ad1 = (const float*)d_in[8];
    const float* b1  = (const float*)d_in[9];
    const float* W2  = (const float*)d_in[10];
    const float* as2 = (const float*)d_in[11];
    const float* ad2 = (const float*)d_in[12];
    const float* b2  = (const float*)d_in[13];
    const float* hW1 = (const float*)d_in[14];
    const float* hb1 = (const float*)d_in[15];
    const float* hW2 = (const float*)d_in[16];
    const float* hb2 = (const float*)d_in[17];

    const int N    = in_sizes[0] / 128;
    const int E    = in_sizes[1] / 2;
    const int Etot = E + N;                 // add_self_loops=True
    const int* srcI = ei;
    const int* dstI = ei + E;

    // --- workspace carve (256B aligned chunks)
    char* ws = (char*)d_ws;
    size_t off = 0;
    auto alloc = [&](size_t bytes) -> float* {
        float* p = (float*)(ws + off);
        off += (bytes + 255) & ~(size_t)255;
        return p;
    };
    float* bufH = alloc((size_t)N * 512 * sizeof(float));   // h = in @ W
    float* bufO = alloc((size_t)N * 512 * sizeof(float));   // layer io ping
    float* bufX = alloc((size_t)N * 512 * sizeof(float));   // layer io pong
    float* asrc = alloc((size_t)N * 4 * sizeof(float));
    float* adst = alloc((size_t)N * 4 * sizeof(float));
    float* mbuf = alloc((size_t)N * 4 * sizeof(float));
    float* dbuf = alloc((size_t)N * 4 * sizeof(float));
    float* ebuf = alloc((size_t)Etot * 4 * sizeof(float));

    auto run_layer = [&](const float* in, int Fin, const float* W, int HC, int H,
                         const float* a_s, const float* a_d, const float* b,
                         float* outb) {
        // 128x128 block, 8 waves of 32x64 (2 A-frags x 4 B-frags)
        dim3 gg((N + 127) / 128, HC / 128);
        gemm_bf16_wmma<2, 4, 4, 2><<<gg, 256, 0, stream>>>(in, W, nullptr, bufH,
                                                           N, Fin, HC, 0, 0);

        int nwv = N * H;
        node_alpha<<<(nwv + 7) / 8, 256, 0, stream>>>(bufH, a_s, a_d, N, H, asrc, adst);

        fill_f32<<<(N * H + 255) / 256, 256, 0, stream>>>(mbuf, -__builtin_inff(), N * H);
        hipMemsetAsync(dbuf, 0, (size_t)N * H * sizeof(float), stream);
        hipMemsetAsync(outb, 0, (size_t)N * HC * sizeof(float), stream);

        int tE = Etot * H;
        edge_logits   <<<(tE + 255) / 256, 256, 0, stream>>>(asrc, adst, srcI, dstI,
                                                             E, Etot, H, ebuf, mbuf);
        edge_norm     <<<(tE + 255) / 256, 256, 0, stream>>>(dstI, E, Etot, H,
                                                             ebuf, mbuf, dbuf);
        edge_aggregate<<<(tE + 7) / 8, 256, 0, stream>>>(bufH, ebuf, dbuf, srcI, dstI,
                                                         E, Etot, H, outb);
        bias_relu<<<((size_t)N * HC + 255) / 256, 256, 0, stream>>>(outb, b, N * HC, HC);
    };

    // layer 0: [N,128] -> [N,512], H=4
    run_layer(x,    128, W0, 512, 4, as0, ad0, b0, bufO);
    // layer 1: [N,512] -> [N,512], H=4
    run_layer(bufO, 512, W1, 512, 4, as1, ad1, b1, bufX);
    // layer 2: [N,512] -> [N,128], H=1 (mean over 1 head == identity)
    run_layer(bufX, 512, W2, 128, 1, as2, ad2, b2, bufO);

    // head: relu([N,128] @ [128,64] + hb1), fused in GEMM (128x64 block shape)
    dim3 gh((N + 127) / 128, 1);
    gemm_bf16_wmma<1, 4, 8, 1><<<gh, 256, 0, stream>>>(bufO, hW1, hb1, bufX,
                                                       N, 128, 64, 1, 1);
    // final [N,64] @ [64,4] + hb2 -> d_out
    head2_k<<<(N * 4 + 255) / 256, 256, 0, stream>>>(bufX, hW2, hb2, N, (float*)d_out);
}
